// BahdanauAttention_89163521065568
// MI455X (gfx1250) — matrix operand order repaired and checked by reading, not compiled, not run on hardware
//
#include <hip/hip_runtime.h>
#include <hip/hip_bf16.h>
#include <math.h>

// ---------------------------------------------------------------------------
// Bahdanau attention, MI455X (gfx1250, wave32, WMMA)
// B=32, T=2048, D=U=512
// ---------------------------------------------------------------------------

#define B_DIM 32
#define T_DIM 2048
#define D_DIM 512
#define U_DIM 512
#define TILE_T 32
#define LDS_STRIDE 520   // 512 + 8 bf16 pad to spread LDS banks

typedef __attribute__((ext_vector_type(16))) __bf16 v16bf;
typedef __attribute__((ext_vector_type(8)))  __bf16 v8bf;
typedef __attribute__((ext_vector_type(4)))  __bf16 v4bf;
typedef __attribute__((ext_vector_type(8)))  float  v8f;

// ---------------------------------------------------------------------------
// Kernel 1a: W2_w [D,U] fp32  ->  W2T [U,D] bf16 (lands in L2, 512 KB)
// ---------------------------------------------------------------------------
__global__ __launch_bounds__(256)
void prep_w2t_kernel(const float* __restrict__ w2, unsigned short* __restrict__ w2t_us)
{
    __bf16* w2t = reinterpret_cast<__bf16*>(w2t_us);
    int idx = blockIdx.x * 256 + threadIdx.x;   // 0 .. 512*512-1, u-major
    int u = idx >> 9;
    int d = idx & 511;
    w2t[idx] = (__bf16)w2[d * U_DIM + u];
}

// ---------------------------------------------------------------------------
// Kernel 1b: qb[b,u] = query[b,:] @ W1_w[:,u] + W1_b[u] + W2_b[u]
// ---------------------------------------------------------------------------
__global__ __launch_bounds__(256)
void prep_qb_kernel(const float* __restrict__ query, const float* __restrict__ w1,
                    const float* __restrict__ w1b, const float* __restrict__ w2b,
                    float* __restrict__ qb)
{
    int idx = blockIdx.x * 256 + threadIdx.x;   // 0 .. 32*512-1
    int b = idx >> 9;
    int u = idx & 511;
    float s = w1b[u] + w2b[u];
    const float* q = query + b * D_DIM;
    #pragma unroll 8
    for (int d = 0; d < D_DIM; ++d)
        s += q[d] * w1[d * U_DIM + u];
    qb[idx] = s;
}

// ---------------------------------------------------------------------------
// Kernel 2 (main): score[b,t] = V_w . tanh(qb[b,:] + values[b,t,:] @ W2) + V_b
// One WG = (b, 32 t-rows). 8 waves: wave&1 -> M-tile, wave>>1 -> N-tile set.
// ---------------------------------------------------------------------------
__global__ __launch_bounds__(256)
void bahdanau_score_kernel(const float* __restrict__ values,
                           const unsigned short* __restrict__ w2t_us,
                           const float* __restrict__ qb,
                           const float* __restrict__ v_w,
                           const float* __restrict__ v_b,
                           float* __restrict__ score)
{
    const __bf16* __restrict__ w2t = reinterpret_cast<const __bf16*>(w2t_us);

    __shared__ __bf16 tileA[TILE_T * LDS_STRIDE];
    __shared__ float  score_lds[TILE_T];

    const int tid = threadIdx.x;
    const int b   = blockIdx.x / (T_DIM / TILE_T);
    const int t0  = (blockIdx.x % (T_DIM / TILE_T)) * TILE_T;

    if (tid < TILE_T) score_lds[tid] = v_b[0];

    // Stage values tile [32 x 512] fp32 -> bf16 in LDS
    const float* vbase = values + (size_t)b * T_DIM * D_DIM + (size_t)t0 * D_DIM;
    #pragma unroll
    for (int i = 0; i < 16; ++i) {
        int idx = tid + i * 256;          // float4 slot id, 0..4095
        int row = idx >> 7;               // /128 float4 per row
        int c4  = idx & 127;
        const float4 v = ((const float4*)(vbase + row * D_DIM))[c4];
        v4bf pk = { (__bf16)v.x, (__bf16)v.y, (__bf16)v.z, (__bf16)v.w };
        *(v4bf*)(&tileA[row * LDS_STRIDE + c4 * 4]) = pk;
    }
    __syncthreads();

    const int wave  = tid >> 5;
    const int lane  = tid & 31;
    const int mtile = wave & 1;           // which 16 rows
    const int wn    = wave >> 1;          // base N-tile (0..3), step 4
    const int m     = mtile * 16 + (lane & 15);
    const int g     = lane >> 4;          // K-half select

    v8f acc[8] = {};                      // 8 N-tiles of 16x16 f32

    // A fragment base: lane layout per ISA 16-bit A 16x32
    const __bf16* aptr = &tileA[m * LDS_STRIDE + g * 8];
    // B fragment base column for this lane
    const int ncol = lane & 15;

    for (int kc = 0; kc < D_DIM; kc += 32) {
        v8bf alo = *(const v8bf*)(aptr + kc);        // K = kc + g*8 .. +7
        v8bf ahi = *(const v8bf*)(aptr + kc + 16);   // K = kc+16+g*8 .. +7
        v16bf a = __builtin_shufflevector(alo, ahi,
                    0,1,2,3,4,5,6,7,8,9,10,11,12,13,14,15);
        #pragma unroll
        for (int j = 0; j < 8; ++j) {
            const int n = wn + 4 * j;
            const __bf16* bp = w2t + (size_t)(n * 16 + ncol) * D_DIM + kc + g * 8;
            if (kc + 32 < D_DIM)
                __builtin_prefetch(bp + 32, 0, 1);   // next K-chunk (global_prefetch)
            v8bf blo = ((const v8bf*)bp)[0];
            v8bf bhi = ((const v8bf*)bp)[2];
            v16bf bb = __builtin_shufflevector(blo, bhi,
                        0,1,2,3,4,5,6,7,8,9,10,11,12,13,14,15);
            acc[j] = __builtin_amdgcn_wmma_f32_16x16x32_bf16(
                         false, a, false, bb, (short)0, acc[j], false, false);
        }
    }

    // Epilogue: tanh + dot with V_w, reduce columns
    float part[8];
    #pragma unroll
    for (int r = 0; r < 8; ++r) part[r] = 0.f;

    #pragma unroll
    for (int j = 0; j < 8; ++j) {
        const int u   = (wn + 4 * j) * 16 + ncol;
        const float qbu = qb[b * U_DIM + u];
        const float vw  = v_w[u];
        #pragma unroll
        for (int r = 0; r < 8; ++r)
            part[r] += tanhf(acc[j][r] + qbu) * vw;
    }

    // Reduce across the 16 column-lanes (keep lane/16 groups separate:
    // they hold different rows). C layout: VGPR r, lane -> M = r + 8*(lane/16).
    #pragma unroll
    for (int mask = 1; mask <= 8; mask <<= 1) {
        #pragma unroll
        for (int r = 0; r < 8; ++r)
            part[r] += __shfl_xor(part[r], mask, 32);
    }

    if (ncol == 0) {
        const int mrow = mtile * 16 + g * 8;
        #pragma unroll
        for (int r = 0; r < 8; ++r)
            atomicAdd(&score_lds[mrow + r], part[r]);   // cross-wave (4 waves/M-tile)
    }
    __syncthreads();

    if (tid < TILE_T)
        score[(size_t)b * T_DIM + t0 + tid] = score_lds[tid];
}

// ---------------------------------------------------------------------------
// Kernel 3: softmax over T per batch -> attn output
// ---------------------------------------------------------------------------
__global__ __launch_bounds__(256)
void softmax_kernel(const float* __restrict__ score, float* __restrict__ attn)
{
    __shared__ float red[256];
    const int b = blockIdx.x;
    const int tid = threadIdx.x;

    float s[8];
    float mx = -3.4e38f;
    #pragma unroll
    for (int i = 0; i < 8; ++i) {
        s[i] = score[b * T_DIM + tid + i * 256];
        mx = fmaxf(mx, s[i]);
    }
    red[tid] = mx;
    __syncthreads();
    for (int off = 128; off > 0; off >>= 1) {
        if (tid < off) red[tid] = fmaxf(red[tid], red[tid + off]);
        __syncthreads();
    }
    mx = red[0];
    __syncthreads();

    float e[8];
    float sum = 0.f;
    #pragma unroll
    for (int i = 0; i < 8; ++i) { e[i] = expf(s[i] - mx); sum += e[i]; }
    red[tid] = sum;
    __syncthreads();
    for (int off = 128; off > 0; off >>= 1) {
        if (tid < off) red[tid] += red[tid + off];
        __syncthreads();
    }
    const float inv = 1.f / red[0];
    #pragma unroll
    for (int i = 0; i < 8; ++i)
        attn[b * T_DIM + tid + i * 256] = e[i] * inv;
}

// ---------------------------------------------------------------------------
// Kernel 4: context[b,d] = sum_t attn[b,t] * values[b,t,d]
// grid = B * (D/64); 4 t-phases per column, LDS combine
// ---------------------------------------------------------------------------
__global__ __launch_bounds__(256)
void context_kernel(const float* __restrict__ values,
                    const float* __restrict__ attn,
                    float* __restrict__ ctx)
{
    __shared__ float red[256];
    const int b     = blockIdx.x >> 3;
    const int d0    = (blockIdx.x & 7) * 64;
    const int tid   = threadIdx.x;
    const int d     = d0 + (tid & 63);
    const int phase = tid >> 6;

    const float* vb = values + (size_t)b * T_DIM * D_DIM;
    const float* ab = attn + b * T_DIM;

    float acc = 0.f;
    for (int t = phase; t < T_DIM; t += 4)
        acc += ab[t] * vb[(size_t)t * D_DIM + d];

    red[tid] = acc;
    __syncthreads();
    if (tid < 64)
        ctx[b * D_DIM + d0 + tid] =
            red[tid] + red[tid + 64] + red[tid + 128] + red[tid + 192];
}

// ---------------------------------------------------------------------------
// Launch
// ---------------------------------------------------------------------------
extern "C" void kernel_launch(void* const* d_in, const int* in_sizes, int n_in,
                              void* d_out, int out_size, void* d_ws, size_t ws_size,
                              hipStream_t stream)
{
    const float* query  = (const float*)d_in[0];   // [B, D]
    const float* values = (const float*)d_in[1];   // [B, T, D]
    const float* W1_w   = (const float*)d_in[2];   // [D, U]
    const float* W1_b   = (const float*)d_in[3];   // [U]
    const float* W2_w   = (const float*)d_in[4];   // [D, U]
    const float* W2_b   = (const float*)d_in[5];   // [U]
    const float* V_w    = (const float*)d_in[6];   // [U, 1]
    const float* V_b    = (const float*)d_in[7];   // [1]

    float* out_ctx  = (float*)d_out;               // context [B, D]
    float* out_attn = out_ctx + B_DIM * D_DIM;     // attn [B, T, 1]

    // Workspace layout (~832 KB total)
    char* ws = (char*)d_ws;
    unsigned short* w2t = (unsigned short*)ws;                    // 512 KB bf16 [U][D]
    float* qb    = (float*)(ws + 512 * 1024);                     // 64 KB  [B][U]
    float* score = (float*)(ws + 512 * 1024 + 64 * 1024);         // 256 KB [B][T]

    prep_w2t_kernel<<<(U_DIM * D_DIM) / 256, 256, 0, stream>>>(W2_w, w2t);
    prep_qb_kernel<<<(B_DIM * U_DIM) / 256, 256, 0, stream>>>(query, W1_w, W1_b, W2_b, qb);
    bahdanau_score_kernel<<<B_DIM * (T_DIM / TILE_T), 256, 0, stream>>>(
        values, w2t, qb, V_w, V_b, score);
    softmax_kernel<<<B_DIM, 256, 0, stream>>>(score, out_attn);
    context_kernel<<<B_DIM * (D_DIM / 64), 256, 0, stream>>>(values, out_attn, out_ctx);
}
